// RenderedStampHead_5669356831149
// MI455X (gfx1250) — matrix operand order repaired and checked by reading, not compiled, not run on hardware
//
#include <hip/hip_runtime.h>
#include <hip/hip_bf16.h>

// ---------------------------------------------------------------------------
// Types / WMMA fragment helpers (CDNA5 wave32, v_wmma_f32_16x16x32_f16)
// ---------------------------------------------------------------------------
typedef _Float16 half_t;
typedef __attribute__((ext_vector_type(16))) _Float16 v16h;
typedef __attribute__((ext_vector_type(8)))  float    v8f;
typedef __attribute__((ext_vector_type(4)))  unsigned int uintx4;

__device__ __forceinline__ v8f vzero8() {
  v8f z; for (int k = 0; k < 8; ++k) z[k] = 0.0f; return z;
}

// A (16x32 f16): lane r=lane&15 -> row M, h=lane>>4.
// elems 0..7  = K in [h*8, h*8+8)
// elems 8..15 = K in [16+h*8, 16+h*8+8)
__device__ __forceinline__ v16h frag_a_ld(const half_t* p0, int ldk) {
  int lane = threadIdx.x & 31;
  int r = lane & 15, h = lane >> 4;
  const half_t* p = p0 + (size_t)r * ldk + h * 8;
  union { v16h v; uintx4 q[2]; } u;
  u.q[0] = *(const uintx4*)(p);
  u.q[1] = *(const uintx4*)(p + 16);
  return u.v;
}

// B (32x16 f16): lane r=lane&15 -> col N, h=lane>>4.
//                elems 0..7 = K in [h*8, h*8+8), elems 8..15 = K in [16+h*8, 16+h*8+8)
// Weights stored [N][K] (out-channel major, contiguous in K).
__device__ __forceinline__ v16h frag_b_ld(const half_t* p0, int ldk) {
  int lane = threadIdx.x & 31;
  int c = lane & 15, h = lane >> 4;
  const half_t* p = p0 + (size_t)c * ldk + h * 8;
  union { v16h v; uintx4 q[2]; } u;
  u.q[0] = *(const uintx4*)(p);
  u.q[1] = *(const uintx4*)(p + 16);
  return u.v;
}

__device__ __forceinline__ v8f wmma16(v16h a, v16h b, v8f c) {
  return __builtin_amdgcn_wmma_f32_16x16x32_f16(
      /*neg_a=*/false, a, /*neg_b=*/false, b,
      /*c_mod=*/(short)0, c, /*reuse_a=*/false, /*reuse_b=*/false);
}

__device__ __forceinline__ float gelu_f(float x) {
  return 0.5f * x * (1.0f + erff(x * 0.70710678118654752f));
}

// ---------------------------------------------------------------------------
// gfx1250 async global->LDS copy helpers (ASYNCcnt-tracked, TDM-adjacent path)
// ---------------------------------------------------------------------------
__device__ __forceinline__ unsigned lds_off_u32(const void* p) {
  // generic -> LDS(addrspace 3) cast; numeric value = byte offset in workgroup LDS
  return (unsigned)(unsigned long long)(const __attribute__((address_space(3))) char*)p;
}

__device__ __forceinline__ void async_ld_b128(unsigned lds_byte_off,
                                              const void* gbase, unsigned gbyte_off) {
  asm volatile("global_load_async_to_lds_b128 %0, %1, %2"
               :
               : "v"(lds_byte_off), "v"(gbyte_off),
                 "s"((unsigned long long)(size_t)gbase)
               : "memory");
}

__device__ __forceinline__ void wait_async_4() {
  asm volatile("s_wait_asynccnt 0x4" ::: "memory");
}
__device__ __forceinline__ void wait_async_0() {
  asm volatile("s_wait_asynccnt 0x0" ::: "memory");
}

// ---------------------------------------------------------------------------
// Problem constants (fixed by reference setup_inputs)
// ---------------------------------------------------------------------------
#define NSRC 512
#define NB   4
#define S31  31
#define BW11 11
#define IC32 32
#define VISH 1024
#define FH   256
#define PX31 (31*31)   // 961
#define PX11 (11*11)   // 121

// ---------------------------------------------------------------------------
// f32 -> f16 conversion (weight prep)
// ---------------------------------------------------------------------------
__global__ __launch_bounds__(256) void cvt_f16_kernel(const float* __restrict__ src,
                                                      half_t* __restrict__ dst, int n) {
  int i = blockIdx.x * 256 + threadIdx.x;
  if (i < n) dst[i] = (half_t)src[i];
}

// ---------------------------------------------------------------------------
// Window extraction: vis_stem (64,1024,1024) -> sw [n][31][31][64] f16
// ---------------------------------------------------------------------------
__global__ __launch_bounds__(256) void extract_sw_kernel(const float* __restrict__ vis,
                                                         const float* __restrict__ spv,
                                                         half_t* __restrict__ sw) {
  int n = blockIdx.x, wy = blockIdx.y;
  int cx = (int)rintf(spv[n * 2 + 0]);
  int cy = (int)rintf(spv[n * 2 + 1]);
  int gy = min(max(cy + wy - 15, 0), VISH - 1);
  for (int idx = threadIdx.x; idx < S31 * 64; idx += 256) {
    int wx = idx >> 6, c = idx & 63;
    int gx = min(max(cx + wx - 15, 0), VISH - 1);
    sw[(((size_t)n * S31 + wy) * S31 + wx) * 64 + c] =
        (half_t)vis[((size_t)c * VISH + gy) * VISH + gx];
  }
}

// bottleneck (256,256,256) -> bwin [n][11][11][256] f32 (pos scaled by 0.25, clipped)
__global__ __launch_bounds__(256) void extract_bwin_kernel(const float* __restrict__ bot,
                                                           const float* __restrict__ spv,
                                                           float* __restrict__ bwin) {
  int n = blockIdx.x, wy = blockIdx.y;
  float sx = fminf(fmaxf(spv[n * 2 + 0] * 0.25f, 0.0f), (float)(FH - 1));
  float sy = fminf(fmaxf(spv[n * 2 + 1] * 0.25f, 0.0f), (float)(FH - 1));
  int cx = (int)rintf(sx), cy = (int)rintf(sy);
  int gy = min(max(cy + wy - 5, 0), FH - 1);
  for (int idx = threadIdx.x; idx < BW11 * 256; idx += 256) {
    int wx = idx >> 8, c = idx & 255;
    int gx = min(max(cx + wx - 5, 0), FH - 1);
    bwin[(((size_t)n * BW11 + wy) * BW11 + wx) * 256 + c] =
        bot[((size_t)c * FH + gy) * FH + gx];
  }
}

// ---------------------------------------------------------------------------
// Generic 3x3 conv (pad 1) on 31x31 channel-last maps via WMMA implicit GEMM.
// grid = (num_maps, 4 row-bands of 8), block = 256 (8 waves).
// Each wave owns output row (band*8 + wave), both 16-wide x-halves.
// B fragments are preloaded per k-step so the WMMA chain issues back-to-back.
// ---------------------------------------------------------------------------
template <int CIN, int COUT, bool GELU_OUT>
__global__ __launch_bounds__(256) void conv3x3_kernel(const half_t* __restrict__ in,
                                                      const float* __restrict__ w,   // [COUT][CIN][3][3]
                                                      const float* __restrict__ bias,
                                                      half_t* __restrict__ out) {
  constexpr int NS = COUT / 16;
  __shared__ half_t sIn[10 * 34 * CIN];   // rows gy0-1 .. gy0+8, cols -1..32, ch-last
  __shared__ half_t sW[COUT * CIN];       // [o][i] for current tap
  __shared__ float  sB[COUT];

  const int tid  = threadIdx.x;
  const int wave = tid >> 5;
  const int src  = blockIdx.x;
  const int band = blockIdx.y;

  // stage zero-padded input tile
  for (int idx = tid; idx < 10 * 34 * CIN; idx += 256) {
    int c   = idx % CIN;
    int rem = idx / CIN;
    int lx  = rem % 34;
    int ly  = rem / 34;
    int gy  = band * 8 - 1 + ly;
    int gx  = lx - 1;
    half_t v = (half_t)0.0f;
    if ((unsigned)gy < 31u && (unsigned)gx < 31u)
      v = in[(((size_t)src * S31 + gy) * S31 + gx) * CIN + c];
    sIn[idx] = v;
  }
  if (tid < COUT) sB[tid] = bias[tid];

  v8f acc[2][NS];
  for (int xm = 0; xm < 2; ++xm)
    for (int ns = 0; ns < NS; ++ns) acc[xm][ns] = vzero8();

  for (int t = 0; t < 9; ++t) {
    __syncthreads();   // also covers input staging on t==0
    for (int idx = tid; idx < COUT * CIN; idx += 256) {
      int o = idx / CIN, i = idx % CIN;
      sW[idx] = (half_t)w[(size_t)(o * CIN + i) * 9 + t];
    }
    __syncthreads();
    const int iy  = wave + t / 3;    // LDS row for this wave's output row
    const int ixo = t % 3;           // LDS col offset for this tap
#pragma unroll
    for (int ks = 0; ks < CIN / 32; ++ks) {
      v16h a0 = frag_a_ld(&sIn[((size_t)iy * 34 + ixo +  0) * CIN + ks * 32], CIN);
      v16h a1 = frag_a_ld(&sIn[((size_t)iy * 34 + ixo + 16) * CIN + ks * 32], CIN);
      v16h bfr[NS];
#pragma unroll
      for (int ns = 0; ns < NS; ++ns)
        bfr[ns] = frag_b_ld(&sW[(size_t)(ns * 16) * CIN + ks * 32], CIN);
#pragma unroll
      for (int ns = 0; ns < NS; ++ns) {
        acc[0][ns] = wmma16(a0, bfr[ns], acc[0][ns]);
        acc[1][ns] = wmma16(a1, bfr[ns], acc[1][ns]);
      }
    }
  }

  // epilogue: D layout -> row = i + 8*(lane>=16), col = lane&15
  int gy = band * 8 + wave;
  if (gy < 31) {
    int lane = tid & 31, r = lane & 15, h = lane >> 4;
    for (int xm = 0; xm < 2; ++xm)
      for (int ns = 0; ns < NS; ++ns)
        for (int i = 0; i < 8; ++i) {
          int x = xm * 16 + i + 8 * h;
          if (x < 31) {
            int ch = ns * 16 + r;
            float v = acc[xm][ns][i] + sB[ch];
            if (GELU_OUT) v = gelu_f(v);
            out[(((size_t)src * S31 + gy) * S31 + x) * COUT + ch] = (half_t)v;
          }
        }
  }
}

// ---------------------------------------------------------------------------
// Depthwise 7x7 (pad 3) + channel LayerNorm -> dnorm f16 [n][121][256]
// one block per (src,pixel); 256 threads = 256 channels
// ---------------------------------------------------------------------------
__global__ __launch_bounds__(256) void dwln_kernel(const float* __restrict__ bwin,
                                                   const float* __restrict__ dww,
                                                   const float* __restrict__ dwb,
                                                   const float* __restrict__ lnw,
                                                   const float* __restrict__ lnb,
                                                   half_t* __restrict__ dnorm) {
  int blk = blockIdx.x;
  int src = blk / PX11, px = blk % PX11;
  int y = px / 11, x = px % 11;
  int ch = threadIdx.x;
  float s = dwb[ch];
  for (int ky = 0; ky < 7; ++ky) {
    int iy = y + ky - 3;
    if ((unsigned)iy >= 11u) continue;
    for (int kx = 0; kx < 7; ++kx) {
      int ix = x + kx - 3;
      if ((unsigned)ix >= 11u) continue;
      s += bwin[((size_t)src * PX11 + iy * 11 + ix) * 256 + ch] * dww[ch * 49 + ky * 7 + kx];
    }
  }
  __shared__ float rs[256], rq[256];
  rs[ch] = s; rq[ch] = s * s;
  __syncthreads();
  for (int off = 128; off > 0; off >>= 1) {
    if (ch < off) { rs[ch] += rs[ch + off]; rq[ch] += rq[ch + off]; }
    __syncthreads();
  }
  float mu  = rs[0] * (1.0f / 256.0f);
  float var = rq[0] * (1.0f / 256.0f) - mu * mu;
  float inv = rsqrtf(var + 1e-6f);
  float v = (s - mu) * inv * lnw[ch] + lnb[ch];
  dnorm[((size_t)src * PX11 + px) * 256 + ch] = (half_t)v;
}

// ---------------------------------------------------------------------------
// pw1: [121x256] @ [256x1024]^T + b, gelu -> f16 [n][121][1024]
// grid=(512, 8), block=256. Async double-buffered K-stages (Kc=32):
// stage k+1 streams via global_load_async_to_lds_b128 while stage k computes.
// ---------------------------------------------------------------------------
__global__ __launch_bounds__(256) void gemm_pw1_kernel(const half_t* __restrict__ A,
                                                       const half_t* __restrict__ Bw, // [1024][256] f16
                                                       const float* __restrict__ bias,
                                                       half_t* __restrict__ out) {
  constexpr int KTOT = 256, KC = KTOT / 32;
  int src = blockIdx.x, nb = blockIdx.y;
  int tid = threadIdx.x, wave = tid >> 5;
  __shared__ half_t sA[2][128 * 32];
  __shared__ half_t sB[2][128 * 32];

  // zero-pad M rows 121..127 once; async loads never touch them
  for (int i = tid; i < 2 * 7 * 32; i += 256) {
    int b = i / (7 * 32), j = i % (7 * 32);
    sA[b][121 * 32 + j] = (half_t)0.0f;
  }
  unsigned aoff[2] = { lds_off_u32(&sA[0][0]), lds_off_u32(&sA[1][0]) };
  unsigned boff[2] = { lds_off_u32(&sB[0][0]), lds_off_u32(&sB[1][0]) };
  __syncthreads();

  auto issue = [&](int kc, int buf) {
    // sA: 512 x 16B chunks (rows >= 121 masked off; wave still issues -> uniform counts)
#pragma unroll
    for (int it = 0; it < 2; ++it) {
      int idx = tid + it * 256;
      int row = idx >> 2, kk = idx & 3;
      if (row < PX11)
        async_ld_b128(aoff[buf] + idx * 16, A,
                      (unsigned)(((src * PX11 + row) * KTOT + kc * 32) * 2 + kk * 16));
    }
#pragma unroll
    for (int it = 0; it < 2; ++it) {
      int idx = tid + it * 256;
      int row = idx >> 2, kk = idx & 3;
      async_ld_b128(boff[buf] + idx * 16, Bw,
                    (unsigned)(((nb * 128 + row) * KTOT + kc * 32) * 2 + kk * 16));
    }
  };

  v8f acc[8];
  for (int ns = 0; ns < 8; ++ns) acc[ns] = vzero8();

  issue(0, 0);
  for (int kc = 0; kc < KC; ++kc) {
    int cur = kc & 1;
    if (kc + 1 < KC) {
      issue(kc + 1, cur ^ 1);
      wait_async_4();   // 4 older ops (stage kc) complete; stage kc+1 in flight
    } else {
      wait_async_0();
    }
    __syncthreads();    // all waves' stage-kc data visible in LDS
    v16h a = frag_a_ld(&sA[cur][(size_t)wave * 16 * 32], 32);
    v16h bfr[8];
#pragma unroll
    for (int ns = 0; ns < 8; ++ns) bfr[ns] = frag_b_ld(&sB[cur][(size_t)ns * 16 * 32], 32);
#pragma unroll
    for (int ns = 0; ns < 8; ++ns) acc[ns] = wmma16(a, bfr[ns], acc[ns]);
    __syncthreads();    // all waves done reading buf[cur] before it is refilled
  }

  int lane = tid & 31, r = lane & 15, h = lane >> 4;
  for (int ns = 0; ns < 8; ++ns)
    for (int i = 0; i < 8; ++i) {
      int row = wave * 16 + i + 8 * h;
      if (row < PX11) {
        int col = nb * 128 + ns * 16 + r;
        float v = gelu_f(acc[ns][i] + bias[col]);
        out[((size_t)src * PX11 + row) * 1024 + col] = (half_t)v;
      }
    }
}

// ---------------------------------------------------------------------------
// pw2: [121x1024] @ [1024x256]^T + b, *gamma, +residual(bwin f32) -> f16
// grid=(512, 2), block=256. Same async double-buffered pipeline, KC=32.
// ---------------------------------------------------------------------------
__global__ __launch_bounds__(256) void gemm_pw2_kernel(const half_t* __restrict__ A,
                                                       const half_t* __restrict__ Bw, // [256][1024] f16
                                                       const float* __restrict__ bias,
                                                       const float* __restrict__ gamma,
                                                       const float* __restrict__ resid,
                                                       half_t* __restrict__ out) {
  constexpr int KTOT = 1024, KC = KTOT / 32;
  int src = blockIdx.x, nb = blockIdx.y;
  int tid = threadIdx.x, wave = tid >> 5;
  __shared__ half_t sA[2][128 * 32];
  __shared__ half_t sB[2][128 * 32];

  for (int i = tid; i < 2 * 7 * 32; i += 256) {
    int b = i / (7 * 32), j = i % (7 * 32);
    sA[b][121 * 32 + j] = (half_t)0.0f;
  }
  unsigned aoff[2] = { lds_off_u32(&sA[0][0]), lds_off_u32(&sA[1][0]) };
  unsigned boff[2] = { lds_off_u32(&sB[0][0]), lds_off_u32(&sB[1][0]) };
  __syncthreads();

  auto issue = [&](int kc, int buf) {
#pragma unroll
    for (int it = 0; it < 2; ++it) {
      int idx = tid + it * 256;
      int row = idx >> 2, kk = idx & 3;
      if (row < PX11)
        async_ld_b128(aoff[buf] + idx * 16, A,
                      (unsigned)(((src * PX11 + row) * KTOT + kc * 32) * 2 + kk * 16));
    }
#pragma unroll
    for (int it = 0; it < 2; ++it) {
      int idx = tid + it * 256;
      int row = idx >> 2, kk = idx & 3;
      async_ld_b128(boff[buf] + idx * 16, Bw,
                    (unsigned)(((nb * 128 + row) * KTOT + kc * 32) * 2 + kk * 16));
    }
  };

  v8f acc[8];
  for (int ns = 0; ns < 8; ++ns) acc[ns] = vzero8();

  issue(0, 0);
  for (int kc = 0; kc < KC; ++kc) {
    int cur = kc & 1;
    if (kc + 1 < KC) {
      issue(kc + 1, cur ^ 1);
      wait_async_4();
    } else {
      wait_async_0();
    }
    __syncthreads();
    v16h a = frag_a_ld(&sA[cur][(size_t)wave * 16 * 32], 32);
    v16h bfr[8];
#pragma unroll
    for (int ns = 0; ns < 8; ++ns) bfr[ns] = frag_b_ld(&sB[cur][(size_t)ns * 16 * 32], 32);
#pragma unroll
    for (int ns = 0; ns < 8; ++ns) acc[ns] = wmma16(a, bfr[ns], acc[ns]);
    __syncthreads();
  }

  int lane = tid & 31, r = lane & 15, h = lane >> 4;
  for (int ns = 0; ns < 8; ++ns)
    for (int i = 0; i < 8; ++i) {
      int row = wave * 16 + i + 8 * h;
      if (row < PX11) {
        int col = nb * 128 + ns * 16 + r;
        float v = acc[ns][i] + bias[col];
        v = v * gamma[col] + resid[((size_t)src * PX11 + row) * 256 + col];
        out[((size_t)src * PX11 + row) * 256 + col] = (half_t)v;
      }
    }
}

// ---------------------------------------------------------------------------
// bnp 1x1 conv (256->64) + gelu + gaussian-weighted spatial pool -> bn_vec [n][64]
// grid=512, block=256
// ---------------------------------------------------------------------------
__global__ __launch_bounds__(256) void bnp_kernel(const half_t* __restrict__ A,  // [n][121][256] f16
                                                  const float* __restrict__ bw,  // [64][256]
                                                  const float* __restrict__ bb,
                                                  float* __restrict__ bn_vec) {
  int src = blockIdx.x, tid = threadIdx.x, wave = tid >> 5;
  __shared__ half_t sA[128 * 64];
  __shared__ half_t sBm[64 * 64];
  __shared__ float sg[PX11];
  __shared__ float sVec[64];
  __shared__ float gsum;
  if (tid == 0) gsum = 0.0f;
  if (tid < 64) sVec[tid] = 0.0f;
  __syncthreads();
  if (tid < PX11) {
    int y = tid / 11, x = tid % 11;
    float ly = -1.0f + 0.2f * (float)y;
    float lx = -1.0f + 0.2f * (float)x;
    float gv = __expf(-(lx * lx + ly * ly) / (2.0f * 0.4f * 0.4f));
    sg[tid] = gv;
    atomicAdd(&gsum, gv);
  }
  __syncthreads();
  if (tid < PX11) sg[tid] /= gsum;

  v8f acc[4];
  for (int ns = 0; ns < 4; ++ns) acc[ns] = vzero8();
  for (int kc = 0; kc < 4; ++kc) {
    __syncthreads();
    for (int idx = tid; idx < 128 * 64; idx += 256) {
      int row = idx >> 6, kk = idx & 63;
      sA[idx] = (row < PX11) ? A[((size_t)src * PX11 + row) * 256 + kc * 64 + kk] : (half_t)0.0f;
    }
    for (int idx = tid; idx < 64 * 64; idx += 256) {
      int o = idx >> 6, kk = idx & 63;
      sBm[idx] = (half_t)bw[(size_t)o * 256 + kc * 64 + kk];
    }
    __syncthreads();
#pragma unroll
    for (int ks = 0; ks < 2; ++ks) {
      v16h a = frag_a_ld(&sA[(size_t)wave * 16 * 64 + ks * 32], 64);
      v16h bfr[4];
#pragma unroll
      for (int ns = 0; ns < 4; ++ns)
        bfr[ns] = frag_b_ld(&sBm[(size_t)ns * 16 * 64 + ks * 32], 64);
#pragma unroll
      for (int ns = 0; ns < 4; ++ns) acc[ns] = wmma16(a, bfr[ns], acc[ns]);
    }
  }
  __syncthreads();
  int lane = tid & 31, r = lane & 15, h = lane >> 4;
  for (int ns = 0; ns < 4; ++ns)
    for (int i = 0; i < 8; ++i) {
      int row = wave * 16 + i + 8 * h;
      if (row < PX11) {
        int ch = ns * 16 + r;
        float v = gelu_f(acc[ns][i] + bb[ch]);
        atomicAdd(&sVec[ch], v * sg[row]);
      }
    }
  __syncthreads();
  if (tid < 64) bn_vec[(size_t)src * 64 + tid] = sVec[tid];
}

// ---------------------------------------------------------------------------
// FiLM: cond[84] = [band_embed(16), tile_pos(2), sub_px(2), bn_vec(64)]
// film = film_w @ cond + film_b ; out[0:32]=1+f (gamma), [32:64]=f (beta)
// grid=2048, block=64
// ---------------------------------------------------------------------------
__global__ __launch_bounds__(64) void film_kernel(const float* __restrict__ ppb,
                                                  const float* __restrict__ band_embed,
                                                  const float* __restrict__ bn_vec,
                                                  const float* __restrict__ film_w,
                                                  const float* __restrict__ film_b,
                                                  float* __restrict__ filmout) {
  int nb = blockIdx.x, n = nb >> 2, b = nb & 3;
  __shared__ float sc[84];
  int t = threadIdx.x;
  if (t < 16) sc[t] = band_embed[b * 16 + t];
  if (t == 16 || t == 17) {
    float p = ppb[(size_t)nb * 2 + (t - 16)];
    sc[t] = fminf(fmaxf(p * (1.0f / 1023.0f), 0.0f), 1.0f);   // tile_w-1 = tile_h-1 = 1023
  }
  if (t == 18 || t == 19) {
    float p = ppb[(size_t)nb * 2 + (t - 18)];
    sc[t] = p - rintf(p);
  }
  if (t >= 20) sc[t + 0] = bn_vec[(size_t)n * 64 + (t - 20)];
  __syncthreads();
  float s = film_b[t];
  for (int j = 0; j < 84; ++j) s += film_w[t * 84 + j] * sc[j];
  filmout[(size_t)nb * 64 + t] = (t < IC32) ? (1.0f + s) : s;
}

// ---------------------------------------------------------------------------
// mod = gelu(intrinsic * gamma + beta) -> f16 [nb][31][31][32]
// grid=(2048,31), block=256
// ---------------------------------------------------------------------------
__global__ __launch_bounds__(256) void mod_kernel(const half_t* __restrict__ intrinsic,
                                                  const float* __restrict__ filmout,
                                                  half_t* __restrict__ modb) {
  int nb = blockIdx.x, y = blockIdx.y, n = nb >> 2;
  for (int idx = threadIdx.x; idx < S31 * IC32; idx += 256) {
    int x = idx >> 5, c = idx & 31;
    float g  = filmout[(size_t)nb * 64 + c];
    float bt = filmout[(size_t)nb * 64 + 32 + c];
    float v  = (float)intrinsic[(((size_t)n * S31 + y) * S31 + x) * IC32 + c];
    v = gelu_f(v * g + bt);
    modb[(((size_t)nb * S31 + y) * S31 + x) * IC32 + c] = (half_t)v;
  }
}

// ---------------------------------------------------------------------------
// dec3 (1x1 conv 32->1) + softplus + normalize by spatial sum -> d_out f32
// grid=2048, block=256
// ---------------------------------------------------------------------------
__global__ __launch_bounds__(256) void dec3_kernel(const half_t* __restrict__ o2,
                                                   const float* __restrict__ w3,
                                                   const float* __restrict__ b3,
                                                   float* __restrict__ out) {
  __shared__ float sw3[32];
  __shared__ float sv[PX31];
  __shared__ float red[256];
  int nb = blockIdx.x, tid = threadIdx.x;
  if (tid < 32) sw3[tid] = w3[tid];
  __syncthreads();
  float bias = b3[0];
  float part = 0.0f;
  for (int px = tid; px < PX31; px += 256) {
    const half_t* p = &o2[((size_t)nb * PX31 + px) * IC32];
    float s = bias;
    for (int c = 0; c < 32; ++c) s += (float)p[c] * sw3[c];
    float sp = (s > 20.0f) ? s : log1pf(__expf(s));
    sv[px] = sp;
    part += sp;
  }
  red[tid] = part;
  __syncthreads();
  for (int off = 128; off > 0; off >>= 1) {
    if (tid < off) red[tid] += red[tid + off];
    __syncthreads();
  }
  float inv = 1.0f / fmaxf(red[0], 1e-8f);
  for (int px = tid; px < PX31; px += 256)
    out[(size_t)nb * PX31 + px] = sv[px] * inv;
}

// ---------------------------------------------------------------------------
// Host orchestration
// ---------------------------------------------------------------------------
extern "C" void kernel_launch(void* const* d_in, const int* in_sizes, int n_in,
                              void* d_out, int out_size, void* d_ws, size_t ws_size,
                              hipStream_t stream) {
  (void)in_sizes; (void)n_in; (void)out_size; (void)ws_size;
  const float* bottleneck = (const float*)d_in[0];
  const float* vis_stem   = (const float*)d_in[1];
  const float* spv        = (const float*)d_in[2];
  const float* ppb        = (const float*)d_in[3];
  const float* ic1_w = (const float*)d_in[4];  const float* ic1_b = (const float*)d_in[5];
  const float* ic2_w = (const float*)d_in[6];  const float* ic2_b = (const float*)d_in[7];
  const float* ic3_w = (const float*)d_in[8];  const float* ic3_b = (const float*)d_in[9];
  const float* dw_w  = (const float*)d_in[10]; const float* dw_b  = (const float*)d_in[11];
  const float* ln_w  = (const float*)d_in[12]; const float* ln_b  = (const float*)d_in[13];
  const float* pw1_w = (const float*)d_in[14]; const float* pw1_b = (const float*)d_in[15];
  const float* pw2_w = (const float*)d_in[16]; const float* pw2_b = (const float*)d_in[17];
  const float* cgam  = (const float*)d_in[18];
  const float* bnp_w = (const float*)d_in[19]; const float* bnp_b = (const float*)d_in[20];
  const float* bemb  = (const float*)d_in[21];
  const float* flm_w = (const float*)d_in[22]; const float* flm_b = (const float*)d_in[23];
  const float* d1_w  = (const float*)d_in[24]; const float* d1_b  = (const float*)d_in[25];
  const float* d2_w  = (const float*)d_in[26]; const float* d2_b  = (const float*)d_in[27];
  const float* d3_w  = (const float*)d_in[28]; const float* d3_b  = (const float*)d_in[29];
  float* out = (float*)d_out;

  // ---- workspace layout (bytes, 256-aligned) ----
  char* ws = (char*)d_ws;
  size_t off = 0;
  auto alloc = [&](size_t bytes) -> char* {
    char* p = ws + off;
    off = (off + bytes + 255) & ~(size_t)255;
    return p;
  };
  half_t* h1     = (half_t*)alloc((size_t)NSRC * PX31 * 64 * 2);        // ic1 out
  half_t* intr   = (half_t*)alloc((size_t)NSRC * PX31 * 32 * 2);        // intrinsic
  float*  bwin   = (float*)alloc((size_t)NSRC * PX11 * 256 * 4);        // bottleneck windows
  half_t* dnorm  = (half_t*)alloc((size_t)NSRC * PX11 * 256 * 2);       // post dw+LN
  half_t* big1   = (half_t*)alloc((size_t)NSRC * PX11 * 1024 * 2);      // pw1_out, later o1
  half_t* bwin2  = (half_t*)alloc((size_t)NSRC * PX11 * 256 * 2);       // post-residual
  float*  bn_vec = (float*)alloc((size_t)NSRC * 64 * 4);
  float*  filmo  = (float*)alloc((size_t)NSRC * NB * 64 * 4);
  half_t* big2   = (half_t*)alloc((size_t)NSRC * NB * PX31 * 32 * 2);   // sw -> h2 -> mod -> o2
  half_t* pw1_wh = (half_t*)alloc((size_t)1024 * 256 * 2);
  half_t* pw2_wh = (half_t*)alloc((size_t)256 * 1024 * 2);

  half_t* sw  = big2;   // [512][31][31][64]  (fits: 512*961*64 <= 2048*961*32)
  half_t* h2  = big2;
  half_t* mod = big2;
  half_t* o1  = big1;
  half_t* o2  = big2;

  // ---- pipeline ----
  cvt_f16_kernel<<<dim3((1024 * 256 + 255) / 256), 256, 0, stream>>>(pw1_w, pw1_wh, 1024 * 256);
  cvt_f16_kernel<<<dim3((256 * 1024 + 255) / 256), 256, 0, stream>>>(pw2_w, pw2_wh, 256 * 1024);

  extract_sw_kernel<<<dim3(NSRC, S31), 256, 0, stream>>>(vis_stem, spv, sw);

  conv3x3_kernel<64, 64, true ><<<dim3(NSRC, 4), 256, 0, stream>>>(sw, ic1_w, ic1_b, h1);
  conv3x3_kernel<64, 64, true ><<<dim3(NSRC, 4), 256, 0, stream>>>(h1, ic2_w, ic2_b, h2);
  conv3x3_kernel<64, 32, false><<<dim3(NSRC, 4), 256, 0, stream>>>(h2, ic3_w, ic3_b, intr);

  extract_bwin_kernel<<<dim3(NSRC, BW11), 256, 0, stream>>>(bottleneck, spv, bwin);
  dwln_kernel<<<dim3(NSRC * PX11), 256, 0, stream>>>(bwin, dw_w, dw_b, ln_w, ln_b, dnorm);
  gemm_pw1_kernel<<<dim3(NSRC, 8), 256, 0, stream>>>(dnorm, pw1_wh, pw1_b, big1);
  gemm_pw2_kernel<<<dim3(NSRC, 2), 256, 0, stream>>>(big1, pw2_wh, pw2_b, cgam, bwin, bwin2);
  bnp_kernel<<<dim3(NSRC), 256, 0, stream>>>(bwin2, bnp_w, bnp_b, bn_vec);

  film_kernel<<<dim3(NSRC * NB), 64, 0, stream>>>(ppb, bemb, bn_vec, flm_w, flm_b, filmo);
  mod_kernel<<<dim3(NSRC * NB, S31), 256, 0, stream>>>(intr, filmo, mod);

  conv3x3_kernel<32, 32, true><<<dim3(NSRC * NB, 4), 256, 0, stream>>>(mod, d1_w, d1_b, o1);
  conv3x3_kernel<32, 32, true><<<dim3(NSRC * NB, 4), 256, 0, stream>>>(o1, d2_w, d2_b, o2);
  dec3_kernel<<<dim3(NSRC * NB), 256, 0, stream>>>(o2, d3_w, d3_b, out);
}
